// SelectiveCopyingModel_40716289966096
// MI455X (gfx1250) — matrix operand order repaired and checked by reading, not compiled, not run on hardware
//
#include <hip/hip_runtime.h>

// ---------------------------------------------------------------------------
// MinGRU model for MI455X (gfx1250): bf16 WMMA GEMMs with double-buffered
// async global->LDS staging (ASYNCcnt), fused linear-space recurrence scan.
// ---------------------------------------------------------------------------

typedef __attribute__((ext_vector_type(16))) __bf16       v16bf;
typedef __attribute__((ext_vector_type(8)))  float        v8f;
typedef __attribute__((ext_vector_type(4)))  unsigned int u32x4;
typedef __attribute__((ext_vector_type(4)))  int          i32x4;
typedef __attribute__((ext_vector_type(2)))  unsigned int u32x2;
typedef __attribute__((ext_vector_type(4)))  float        f32x4;

#define BB    16
#define TT    4096
#define EE    1024
#define HH    1024
#define VV    2048
#define MTOT  (BB * TT)         // 65536 rows
#define NMEM  16

#define AS1 __attribute__((address_space(1)))
#define AS3 __attribute__((address_space(3)))

static __device__ __forceinline__ __bf16 f2bf(float f) {
    unsigned u = __builtin_bit_cast(unsigned, f);
    u += 0x7FFFu + ((u >> 16) & 1u);               // round-to-nearest-even
    unsigned short s = (unsigned short)(u >> 16);
    return __builtin_bit_cast(__bf16, s);
}

union FragAB { v16bf v; u32x4 q[2]; };

// --- CDNA5 async global->LDS copy (16B per lane), tracked by ASYNCcnt ------
static __device__ __forceinline__ void async_cp16(const __bf16* g, __bf16* l) {
#if __has_builtin(__builtin_amdgcn_global_load_async_to_lds_b128)
    __builtin_amdgcn_global_load_async_to_lds_b128(
        (AS1 i32x4*)g, (AS3 i32x4*)l, 0, 0);
#else
    asm volatile("global_load_async_to_lds_b128 %0, %1, off"
                 :: "v"((unsigned)(size_t)(AS3 void*)l), "v"(g)
                 : "memory");
#endif
}

template <int N>
static __device__ __forceinline__ void wait_async_le() {
#if __has_builtin(__builtin_amdgcn_s_wait_asynccnt)
    __builtin_amdgcn_s_wait_asynccnt((unsigned short)N);
#else
    asm volatile("s_wait_asynccnt %0" :: "i"(N) : "memory");
#endif
}

// ---------------------------------------------------------------------------
// fp32 -> bf16 weight cast (4 floats / thread)
// ---------------------------------------------------------------------------
__global__ void cast_f32_bf16(const float* __restrict__ s, __bf16* __restrict__ d, int n4) {
    int i = blockIdx.x * 256 + threadIdx.x;
    if (i >= n4) return;
    f32x4 v = *(const f32x4*)(s + (size_t)i * 4);
    u32x2 p;
    p[0] = (unsigned)__builtin_bit_cast(unsigned short, f2bf(v[0]))
         | ((unsigned)__builtin_bit_cast(unsigned short, f2bf(v[1])) << 16);
    p[1] = (unsigned)__builtin_bit_cast(unsigned short, f2bf(v[2]))
         | ((unsigned)__builtin_bit_cast(unsigned short, f2bf(v[3])) << 16);
    *(u32x2*)(d + (size_t)i * 4) = p;
}

// ---------------------------------------------------------------------------
// Embedding gather: h_f32 = emb[x], h_bf16 = bf16(h_f32)
// ---------------------------------------------------------------------------
__global__ void embed_gather_cast(const int* __restrict__ x, const float* __restrict__ emb,
                                  float* __restrict__ hf, __bf16* __restrict__ hb) {
    int idx = blockIdx.x * 256 + threadIdx.x;   // one float4 chunk
    int m = idx >> 8;                           // 1024/4 = 256 chunks per row
    int c = (idx & 255) * 4;
    int tok = x[m];
    f32x4 v = *(const f32x4*)(emb + (size_t)tok * EE + c);
    *(f32x4*)(hf + (size_t)m * EE + c) = v;
    u32x2 p;
    p[0] = (unsigned)__builtin_bit_cast(unsigned short, f2bf(v[0]))
         | ((unsigned)__builtin_bit_cast(unsigned short, f2bf(v[1])) << 16);
    p[1] = (unsigned)__builtin_bit_cast(unsigned short, f2bf(v[2]))
         | ((unsigned)__builtin_bit_cast(unsigned short, f2bf(v[3])) << 16);
    *(u32x2*)(hb + (size_t)m * EE + c) = p;
}

// ---------------------------------------------------------------------------
// bf16 WMMA GEMM: out(M,N) = A(M,K) * W(N,K)^T + bias
// 128x128 tile per workgroup, 8 waves (wave32), each wave = 64x32 via
// 4x2 accumulators of v_wmma_f32_16x16x32_bf16.
// Double-buffered LDS filled by global_load_async_to_lds_b128 (ASYNCcnt);
// tile i+1 copies overlap tile i WMMAs. K step 32.
// REMAP gathers A rows b*TT + (TT-16) + tt for the final 16-timestep logits.
// ---------------------------------------------------------------------------
template <bool REMAP>
__global__ __launch_bounds__(256, 1)
void gemm_bf16_wmma(const __bf16* __restrict__ A, const __bf16* __restrict__ W,
                    const float* __restrict__ bias, float* __restrict__ out,
                    int K, int N) {
    __shared__ __bf16 sA[2][128 * 40];   // row stride 40 halves = 80 B (16B-aligned)
    __shared__ __bf16 sB[2][128 * 40];

    const int tid  = threadIdx.x;
    const int n0   = blockIdx.x * 128;
    const int m0   = blockIdx.y * 128;
    const int wave = tid >> 5;            // wave32
    const int lane = tid & 31;
    const int wm   = wave >> 2;           // 0..1  (64 rows each)
    const int wn   = wave & 3;            // 0..3  (32 cols each)
    const int ml   = lane & 15;
    const int half = lane >> 4;

    // this thread's two 16B staging chunks (c and c+256 of 512)
    const int c0_row = tid >> 2;
    const int cc0    = (tid & 3) * 8;
    const int c1_row = (tid + 256) >> 2;
    const int cc1    = ((tid + 256) & 3) * 8;
    int arow0 = m0 + c0_row, arow1 = m0 + c1_row;
    if (REMAP) {
        arow0 = (arow0 >> 4) * TT + (TT - NMEM) + (arow0 & 15);
        arow1 = (arow1 >> 4) * TT + (TT - NMEM) + (arow1 & 15);
    }
    const __bf16* gA0 = A + (size_t)arow0 * K + cc0;
    const __bf16* gA1 = A + (size_t)arow1 * K + cc1;
    const __bf16* gB0 = W + (size_t)(n0 + c0_row) * K + cc0;
    const __bf16* gB1 = W + (size_t)(n0 + c1_row) * K + cc1;

    v8f acc[4][2];
    #pragma unroll
    for (int s = 0; s < 4; ++s)
        #pragma unroll
        for (int t = 0; t < 2; ++t)
            #pragma unroll
            for (int e = 0; e < 8; ++e) acc[s][t][e] = 0.0f;

    auto issue_tile = [&](int kb, int buf) {
        async_cp16(gA0 + kb, &sA[buf][c0_row * 40 + cc0]);
        async_cp16(gA1 + kb, &sA[buf][c1_row * 40 + cc1]);
        async_cp16(gB0 + kb, &sB[buf][c0_row * 40 + cc0]);
        async_cp16(gB1 + kb, &sB[buf][c1_row * 40 + cc1]);
    };

    auto compute_tile = [&](int buf) {
        FragAB aF[4], bF[2];
        #pragma unroll
        for (int s = 0; s < 4; ++s) {
            const __bf16* base = &sA[buf][(wm * 64 + s * 16 + ml) * 40];
            aF[s].q[0] = *(const u32x4*)(base + half * 8);
            aF[s].q[1] = *(const u32x4*)(base + 16 + half * 8);
        }
        #pragma unroll
        for (int t = 0; t < 2; ++t) {
            const __bf16* base = &sB[buf][(wn * 32 + t * 16 + ml) * 40];
            bF[t].q[0] = *(const u32x4*)(base + half * 8);
            bF[t].q[1] = *(const u32x4*)(base + 16 + half * 8);
        }
        #pragma unroll
        for (int s = 0; s < 4; ++s)
            #pragma unroll
            for (int t = 0; t < 2; ++t)
                acc[s][t] = __builtin_amdgcn_wmma_f32_16x16x32_bf16(
                    false, aF[s].v, false, bF[t].v, (short)0, acc[s][t], false, false);
    };

    const int nk = K / 32;
    issue_tile(0, 0);
    for (int i = 0; i < nk - 1; ++i) {
        issue_tile((i + 1) * 32, (i + 1) & 1);   // overlap next copy with compute
        wait_async_le<4>();                      // tile i resident (in-order)
        __syncthreads();
        compute_tile(i & 1);
        __syncthreads();                         // protect buf before re-issue
    }
    wait_async_le<0>();
    __syncthreads();
    compute_tile((nk - 1) & 1);

    // epilogue: C/D layout — element r lives at M = r + 8*half, N = ml
    #pragma unroll
    for (int t = 0; t < 2; ++t) {
        int ncol = n0 + wn * 32 + t * 16 + ml;
        float bv = bias[ncol];
        #pragma unroll
        for (int s = 0; s < 4; ++s) {
            #pragma unroll
            for (int r = 0; r < 8; ++r) {
                int grow = m0 + wm * 64 + s * 16 + r + 8 * half;
                out[(size_t)grow * N + ncol] = acc[s][t][r] + bv;
            }
        }
    }
}

// ---------------------------------------------------------------------------
// minGRU recurrence (linear space, equivalent to the log-space scan since all
// terms are positive): h_t = (1-z)h + z*g(h~), fused residual, dual-precision
// output. One lane per (batch, channel) — 16384 lanes, coalesced per step.
// ---------------------------------------------------------------------------
__global__ void mingru_scan(const float* __restrict__ gh, float* __restrict__ hf,
                            __bf16* __restrict__ hb) {
    int b = blockIdx.x >> 2;                       // 4 blocks per batch (1024/256)
    int j = (blockIdx.x & 3) * 256 + threadIdx.x;  // channel
    float h = 0.5f;                                // h0 = g(0)
    size_t i2 = (size_t)b * TT * (2 * HH) + j;
    size_t i1 = (size_t)b * TT * HH + j;
    for (int t = 0; t < TT; ++t) {
        float gate = gh[i2];
        float hid  = gh[i2 + HH];
        float z  = 1.0f / (1.0f + __expf(-gate));
        float gt = (hid >= 0.0f) ? (hid + 0.5f) : (1.0f / (1.0f + __expf(-hid)));
        h = h + z * (gt - h);
        float o = h + hf[i1];                      // residual
        hf[i1] = o;
        hb[i1] = f2bf(o);
        i2 += 2 * HH;
        i1 += HH;
    }
}

// ---------------------------------------------------------------------------
extern "C" void kernel_launch(void* const* d_in, const int* in_sizes, int n_in,
                              void* d_out, int out_size, void* d_ws, size_t ws_size,
                              hipStream_t stream) {
    (void)in_sizes; (void)n_in; (void)out_size;
    const int*   x   = (const int*)d_in[0];
    const float* emb = (const float*)d_in[1];
    const float* W0  = (const float*)d_in[2];
    const float* b0  = (const float*)d_in[3];
    const float* W1  = (const float*)d_in[4];
    const float* b1  = (const float*)d_in[5];
    const float* Wl  = (const float*)d_in[6];
    const float* bl  = (const float*)d_in[7];
    float* out = (float*)d_out;

    const size_t sz_hf = (size_t)MTOT * EE * 4;       // 268 MB
    const size_t sz_hb = (size_t)MTOT * EE * 2;       // 134 MB
    const size_t sz_gh = (size_t)MTOT * (2 * HH) * 4; // 537 MB
    const size_t sz_w  = (size_t)VV * EE * 2;         // 4 MB each
    if (ws_size < sz_hf + sz_hb + sz_gh + 3 * sz_w) return;

    char* ws = (char*)d_ws;
    float*  hf  = (float*)ws;   ws += sz_hf;
    __bf16* hb  = (__bf16*)ws;  ws += sz_hb;
    float*  gh  = (float*)ws;   ws += sz_gh;
    __bf16* W0b = (__bf16*)ws;  ws += sz_w;
    __bf16* W1b = (__bf16*)ws;  ws += sz_w;
    __bf16* Wlb = (__bf16*)ws;

    const int wn4 = (VV * EE) / 4;
    cast_f32_bf16<<<wn4 / 256, 256, 0, stream>>>(W0, W0b, wn4);
    cast_f32_bf16<<<wn4 / 256, 256, 0, stream>>>(W1, W1b, wn4);
    cast_f32_bf16<<<wn4 / 256, 256, 0, stream>>>(Wl, Wlb, wn4);

    embed_gather_cast<<<(MTOT * EE / 4) / 256, 256, 0, stream>>>(x, emb, hf, hb);

    dim3 gemm_grid(VV / 128, MTOT / 128);    // 16 x 512
    gemm_bf16_wmma<false><<<gemm_grid, 256, 0, stream>>>(hb, W0b, b0, gh, EE, 2 * HH);
    mingru_scan<<<BB * (HH / 256), 256, 0, stream>>>(gh, hf, hb);

    gemm_bf16_wmma<false><<<gemm_grid, 256, 0, stream>>>(hb, W1b, b1, gh, HH, 2 * HH);
    mingru_scan<<<BB * (HH / 256), 256, 0, stream>>>(gh, hf, hb);

    // final projection: only the last 16 timesteps (256 gathered rows)
    dim3 logits_grid(VV / 128, (BB * NMEM) / 128);   // 16 x 2
    gemm_bf16_wmma<true><<<logits_grid, 256, 0, stream>>>(hb, Wlb, bl, out, HH, VV);
}
